// InternImageLayer_2018634629709
// MI455X (gfx1250) — compile-verified, matrix operand order checked by reading, not run on hardware
//
#include <hip/hip_runtime.h>
#include <hip/hip_bf16.h>

typedef _Float16 v8h  __attribute__((ext_vector_type(8)));
typedef _Float16 v16h __attribute__((ext_vector_type(16)));
typedef float    v8f  __attribute__((ext_vector_type(8)));

static constexpr int TOK = 25088;   // 8*56*56
static constexpr int HID = 1024;
static constexpr int HP  = 58;      // padded H/W

// ---------------------------------------------------------------------------
// Weight convert: W[K][Nt] f32 (row-major)  ->  Wt[Nt][K] f16 (column-major)
// ---------------------------------------------------------------------------
__global__ void w_transpose_f16(const float* __restrict__ W,
                                _Float16* __restrict__ Wt, int K, int Nt) {
  int idx = blockIdx.x * blockDim.x + threadIdx.x;
  if (idx >= K * Nt) return;
  int nt = idx / K, k = idx % K;
  Wt[idx] = (_Float16)W[(size_t)k * Nt + nt];
}

// ---------------------------------------------------------------------------
// LayerNorm over C=256, one token per 256-thread block, f16 output
// ---------------------------------------------------------------------------
__global__ __launch_bounds__(256) void ln256(const float* __restrict__ x,
                                             const float* __restrict__ g,
                                             const float* __restrict__ b,
                                             _Float16* __restrict__ outH) {
  __shared__ float red[256];
  const int t = threadIdx.x;
  const size_t base = (size_t)blockIdx.x * 256;
  float v = x[base + t];
  red[t] = v; __syncthreads();
  for (int s = 128; s > 0; s >>= 1) { if (t < s) red[t] += red[t + s]; __syncthreads(); }
  float mu = red[0] * (1.0f / 256.0f);
  __syncthreads();
  float d = v - mu;
  red[t] = d * d; __syncthreads();
  for (int s = 128; s > 0; s >>= 1) { if (t < s) red[t] += red[t + s]; __syncthreads(); }
  float var = red[0] * (1.0f / 256.0f);
  float y = d * rsqrtf(var + 1e-6f) * g[t] + b[t];
  outH[base + t] = (_Float16)y;
}

// ---------------------------------------------------------------------------
// Depthwise 3x3 conv + bias + LayerNorm + exact GELU, fused per token
// ---------------------------------------------------------------------------
__global__ __launch_bounds__(256) void dwconv_ln_gelu(
    const _Float16* __restrict__ xin, const float* __restrict__ dww,
    const float* __restrict__ dwb, const float* __restrict__ g,
    const float* __restrict__ b, _Float16* __restrict__ outH) {
  __shared__ float red[256];
  const int c = threadIdx.x;
  const int token = blockIdx.x;
  const int n = token / 3136, rem = token % 3136;
  const int h = rem / 56, w = rem % 56;
  float acc = dwb[c];
#pragma unroll
  for (int ky = 0; ky < 3; ++ky) {
    const int yy = h + ky - 1;
    if (yy < 0 || yy >= 56) continue;          // uniform per block
#pragma unroll
    for (int kx = 0; kx < 3; ++kx) {
      const int xx = w + kx - 1;
      if (xx < 0 || xx >= 56) continue;        // uniform per block
      acc += dww[c * 9 + ky * 3 + kx] *
             (float)xin[(((size_t)n * 56 + yy) * 56 + xx) * 256 + c];
    }
  }
  red[c] = acc; __syncthreads();
  for (int s = 128; s > 0; s >>= 1) { if (c < s) red[c] += red[c + s]; __syncthreads(); }
  float mu = red[0] * (1.0f / 256.0f);
  __syncthreads();
  float d = acc - mu;
  red[c] = d * d; __syncthreads();
  for (int s = 128; s > 0; s >>= 1) { if (c < s) red[c] += red[c + s]; __syncthreads(); }
  float var = red[0] * (1.0f / 256.0f);
  float y = d * rsqrtf(var + 1e-6f) * g[c] + b[c];
  y = 0.5f * y * (1.0f + erff(y * 0.70710678118654752f));
  outH[(size_t)token * 256 + c] = (_Float16)y;
}

// ---------------------------------------------------------------------------
// Softmax over P=9 points; one thread per (token, group) row
// ---------------------------------------------------------------------------
__global__ void softmax9(float* __restrict__ m, int rows) {
  int idx = blockIdx.x * blockDim.x + threadIdx.x;
  if (idx >= rows) return;
  size_t base = (size_t)idx * 9;
  float v[9]; float mx = -1e30f;
#pragma unroll
  for (int p = 0; p < 9; ++p) { v[p] = m[base + p]; mx = fmaxf(mx, v[p]); }
  float s = 0.f;
#pragma unroll
  for (int p = 0; p < 9; ++p) { v[p] = expf(v[p] - mx); s += v[p]; }
  float inv = 1.0f / s;
#pragma unroll
  for (int p = 0; p < 9; ++p) m[base + p] = v[p] * inv;
}

// ---------------------------------------------------------------------------
// DCNv3 deformable bilinear sampling + mask-weighted accumulation
// thread = (group, channel) of one token; xp is f16 padded NHWC (58x58)
// ---------------------------------------------------------------------------
__global__ __launch_bounds__(256) void dcn_sample(
    const _Float16* __restrict__ xp, const float* __restrict__ off,
    const float* __restrict__ msk, _Float16* __restrict__ outH) {
  const int token = blockIdx.x;
  const int t = threadIdx.x;
  const int gidx = t >> 4, c = t & 15;
  const int n = token / 3136, rem = token % 3136;
  const int h = rem / 56, w = rem % 56;
  const int ch = gidx * 16 + c;
  const _Float16* img = xp + (size_t)n * (HP * HP) * 256 + ch;
  const float* op = off + (size_t)token * 288 + gidx * 18;
  const float* mp = msk + (size_t)token * 144 + gidx * 9;
  float acc = 0.f;
#pragma unroll
  for (int p = 0; p < 9; ++p) {
    const float gx = (float)(p / 3 - 1);     // x-major flatten (meshgrid 'ij')
    const float gy = (float)(p % 3 - 1);
    const float px = (float)(w + 1) + gx + op[p * 2 + 0];
    const float py = (float)(h + 1) + gy + op[p * 2 + 1];
    const float fx = floorf(px), fy = floorf(py);
    const int x0 = (int)fx, y0 = (int)fy;
    const float lx = px - fx, ly = py - fy;
    float s = 0.f;
#define TAP(Y, X, WGT)                                                        \
    if ((unsigned)(Y) < (unsigned)HP && (unsigned)(X) < (unsigned)HP)         \
      s += (WGT) * (float)img[((size_t)(Y) * HP + (X)) * 256];
    TAP(y0,     x0,     (1.f - lx) * (1.f - ly))
    TAP(y0,     x0 + 1, lx * (1.f - ly))
    TAP(y0 + 1, x0,     (1.f - lx) * ly)
    TAP(y0 + 1, x0 + 1, lx * ly)
#undef TAP
    acc += mp[p] * s;
  }
  outH[(size_t)token * 256 + ch] = (_Float16)acc;
}

// ---------------------------------------------------------------------------
// Fragment load helper: two contiguous 16B loads per lane per the CDNA5
// 16-bit A/B VGPR layout (kb = 8 for lanes 16..31).
// ---------------------------------------------------------------------------
__device__ __forceinline__ v16h ldfrag(const _Float16* __restrict__ base,
                                       int k, int kb) {
  v8h lo = *(const v8h*)(base + k + kb);        // K = kb..kb+7
  v8h hi = *(const v8h*)(base + k + kb + 16);   // K = 16+kb..16+kb+7
  return __builtin_shufflevector(lo, hi, 0, 1, 2, 3, 4, 5, 6, 7,
                                 8, 9, 10, 11, 12, 13, 14, 15);
}

// ---------------------------------------------------------------------------
// WMMA GEMM, register-blocked: each wave computes a (16*MT) x (16*NT) tile
// (MT*NT accumulators), reusing each B fragment across MT tiles and each A
// fragment across NT tiles.  Block = 8 waves arranged (8/WN) in M x WN in N.
// EPI: 0 = f32 store, 1 = GELU -> f16 store, 2 = res + scale*(acc+bias) f32,
//      3 = f16 scatter into zero-padded NHWC (58x58), N must be 256.
// ---------------------------------------------------------------------------
template <int MT, int NT, int WN, int EPI>
__global__ __launch_bounds__(256) void gemm_wmma(
    const _Float16* __restrict__ A, const _Float16* __restrict__ Bt,
    const float* __restrict__ bias, float* __restrict__ outF,
    _Float16* __restrict__ outH, const float* __restrict__ res,
    const float* __restrict__ scale, int M, int N, int K) {
  constexpr int WM = 8 / WN;               // waves along M
  const int wave = threadIdx.x >> 5;
  const int lane = threadIdx.x & 31;
  const int hl = lane & 15;
  const int hi = lane >> 4;
  const int kb = hi * 8;                   // fragment K-base per half-wave
  const int wm = wave % WM, wn = wave / WM;
  const int m0 = (blockIdx.y * WM + wm) * (16 * MT);
  const int n0 = (blockIdx.x * WN + wn) * (16 * NT);

  const _Float16* aR[MT];
  const _Float16* bC[NT];
#pragma unroll
  for (int mi = 0; mi < MT; ++mi) aR[mi] = A + (size_t)(m0 + mi * 16 + hl) * K;
#pragma unroll
  for (int ni = 0; ni < NT; ++ni) bC[ni] = Bt + (size_t)(n0 + ni * 16 + hl) * K;

  v8f acc[MT][NT] = {};
  for (int k = 0; k < K; k += 32) {
    v16h b[NT];
#pragma unroll
    for (int ni = 0; ni < NT; ++ni) b[ni] = ldfrag(bC[ni], k, kb);
#pragma unroll
    for (int mi = 0; mi < MT; ++mi) {
      v16h a = ldfrag(aR[mi], k, kb);
#pragma unroll
      for (int ni = 0; ni < NT; ++ni)
        acc[mi][ni] = __builtin_amdgcn_wmma_f32_16x16x32_f16(
            false, a, false, b[ni], (short)0, acc[mi][ni], false, false);
    }
  }

#pragma unroll
  for (int mi = 0; mi < MT; ++mi) {
#pragma unroll
    for (int ni = 0; ni < NT; ++ni) {
      const v8f a = acc[mi][ni];
      const int col = n0 + ni * 16 + hl;
      const float bv = bias[col];
      const int rbase = m0 + mi * 16 + hi * 8;   // C/D layout: M = v + 8*hi
#pragma unroll
      for (int v = 0; v < 8; ++v) {
        const int row = rbase + v;
        float val = a[v] + bv;
        if (EPI == 0) {
          outF[(size_t)row * N + col] = val;
        } else if (EPI == 1) {
          val = 0.5f * val * (1.0f + erff(val * 0.70710678118654752f));
          outH[(size_t)row * N + col] = (_Float16)val;
        } else if (EPI == 2) {
          outF[(size_t)row * N + col] =
              res[(size_t)row * N + col] + scale[col] * val;
        } else {  // EPI == 3: f16 scatter into padded NHWC, N == 256
          int n = row / 3136, rem = row % 3136;
          int h = rem / 56, w = rem % 56;
          size_t idx = (((size_t)n * HP + (h + 1)) * HP + (w + 1)) * 256 + col;
          outH[idx] = (_Float16)val;
        }
      }
    }
  }
}

// ---------------------------------------------------------------------------
extern "C" void kernel_launch(void* const* d_in, const int* in_sizes, int n_in,
                              void* d_out, int out_size, void* d_ws, size_t ws_size,
                              hipStream_t stream) {
  const float* x      = (const float*)d_in[0];
  const float* n1g    = (const float*)d_in[1];
  const float* n1b    = (const float*)d_in[2];
  const float* n2g    = (const float*)d_in[3];
  const float* n2b    = (const float*)d_in[4];
  const float* gamma1 = (const float*)d_in[5];
  const float* gamma2 = (const float*)d_in[6];
  const float* dw_w   = (const float*)d_in[7];
  const float* dw_b   = (const float*)d_in[8];
  const float* dwlng  = (const float*)d_in[9];
  const float* dwlnb  = (const float*)d_in[10];
  const float* off_w  = (const float*)d_in[11];
  const float* off_b  = (const float*)d_in[12];
  const float* mask_w = (const float*)d_in[13];
  const float* mask_b = (const float*)d_in[14];
  const float* in_w   = (const float*)d_in[15];
  const float* in_b   = (const float*)d_in[16];
  const float* out_w  = (const float*)d_in[17];
  const float* out_b  = (const float*)d_in[18];
  const float* fc1_w  = (const float*)d_in[19];
  const float* fc1_b  = (const float*)d_in[20];
  const float* fc2_w  = (const float*)d_in[21];
  const float* fc2_b  = (const float*)d_in[22];

  char* p = (char*)d_ws;
  auto bump = [&](size_t bytes) -> char* {
    char* r = p;
    p += (bytes + 255) & ~(size_t)255;
    return r;
  };

  const size_t XP_BYTES = (size_t)8 * HP * HP * 256 * 2;    // f16
  _Float16* xp    = (_Float16*)bump(XP_BYTES);              // padded in-proj
  _Float16* xn1   = (_Float16*)bump((size_t)TOK * 256 * 2); // LN1(x) f16
  _Float16* x1h   = (_Float16*)bump((size_t)TOK * 256 * 2); // gelu(ln(dwconv))
  float*    offb  = (float*)bump((size_t)TOK * 288 * 4);    // offsets
  float*    mkb   = (float*)bump((size_t)TOK * 144 * 4);    // mask (softmaxed)
  _Float16* dcn   = (_Float16*)bump((size_t)TOK * 256 * 2); // sampled output
  float*    x2    = (float*)bump((size_t)TOK * 256 * 4);    // x + g1*h
  _Float16* xn2   = (_Float16*)bump((size_t)TOK * 256 * 2); // LN2(x2) f16
  _Float16* hid   = (_Float16*)bump((size_t)TOK * HID * 2); // MLP hidden
  _Float16* inWt  = (_Float16*)bump((size_t)256 * 256 * 2);
  _Float16* offWt = (_Float16*)bump((size_t)288 * 256 * 2);
  _Float16* mskWt = (_Float16*)bump((size_t)144 * 256 * 2);
  _Float16* outWt = (_Float16*)bump((size_t)256 * 256 * 2);
  _Float16* fc1Wt = (_Float16*)bump((size_t)HID * 256 * 2);
  _Float16* fc2Wt = (_Float16*)bump((size_t)256 * HID * 2);

  hipMemsetAsync(xp, 0, XP_BYTES, stream);

  auto cdiv = [](int a, int b) { return (a + b - 1) / b; };

  // weights -> f16 column-major [Nout][K]
  w_transpose_f16<<<cdiv(256 * 256, 256), 256, 0, stream>>>(in_w,   inWt,  256, 256);
  w_transpose_f16<<<cdiv(256 * 288, 256), 256, 0, stream>>>(off_w,  offWt, 256, 288);
  w_transpose_f16<<<cdiv(256 * 144, 256), 256, 0, stream>>>(mask_w, mskWt, 256, 144);
  w_transpose_f16<<<cdiv(256 * 256, 256), 256, 0, stream>>>(out_w,  outWt, 256, 256);
  w_transpose_f16<<<cdiv(256 * HID, 256), 256, 0, stream>>>(fc1_w,  fc1Wt, 256, HID);
  w_transpose_f16<<<cdiv(HID * 256, 256), 256, 0, stream>>>(fc2_w,  fc2Wt, HID, 256);

  const dim3 blk(256);
  // wave tile 64x32 (MT=4,NT=2): block 256M x 64N (WN=2) or 512M x 32N (WN=1)

  // 1) LN1
  ln256<<<TOK, blk, 0, stream>>>(x, n1g, n1b, xn1);
  // 2) in-proj GEMM -> padded NHWC f16   (grid 4 x 98)
  gemm_wmma<4, 2, 2, 3><<<dim3(256 / 64, TOK / 256), blk, 0, stream>>>(
      xn1, inWt, in_b, nullptr, xp, nullptr, nullptr, TOK, 256, 256);
  // 3) dwconv + LN + GELU
  dwconv_ln_gelu<<<TOK, blk, 0, stream>>>(xn1, dw_w, dw_b, dwlng, dwlnb, x1h);
  // 4) offset projection (N=288: block 512x32, grid 9 x 49)
  gemm_wmma<4, 2, 1, 0><<<dim3(288 / 32, TOK / 512), blk, 0, stream>>>(
      x1h, offWt, off_b, offb, nullptr, nullptr, nullptr, TOK, 288, 256);
  //    mask projection (N=144: block 512x16, grid 9 x 49)
  gemm_wmma<4, 1, 1, 0><<<dim3(144 / 16, TOK / 512), blk, 0, stream>>>(
      x1h, mskWt, mask_b, mkb, nullptr, nullptr, nullptr, TOK, 144, 256);
  // 5) softmax over 9 points
  softmax9<<<cdiv(TOK * 16, 256), blk, 0, stream>>>(mkb, TOK * 16);
  // 6) deformable sampling (f16 gathers)
  dcn_sample<<<TOK, blk, 0, stream>>>(xp, offb, mkb, dcn);
  // 7) out-proj GEMM + residual: x2 = x + gamma1 * (dcn @ out_w + out_b)
  gemm_wmma<4, 2, 2, 2><<<dim3(256 / 64, TOK / 256), blk, 0, stream>>>(
      dcn, outWt, out_b, x2, nullptr, x, gamma1, TOK, 256, 256);
  // 8) LN2
  ln256<<<TOK, blk, 0, stream>>>(x2, n2g, n2b, xn2);
  // 9) fc1 + GELU -> f16 (grid 16 x 98)
  gemm_wmma<4, 2, 2, 1><<<dim3(HID / 64, TOK / 256), blk, 0, stream>>>(
      xn2, fc1Wt, fc1_b, nullptr, hid, nullptr, nullptr, TOK, HID, 256);
  // 10) fc2 + residual -> final output
  gemm_wmma<4, 2, 2, 2><<<dim3(256 / 64, TOK / 256), blk, 0, stream>>>(
      hid, fc2Wt, fc2_b, (float*)d_out, nullptr, x2, gamma2, TOK, 256, HID);
}
